// MultiHeadSelfAttention_71322226917603
// MI455X (gfx1250) — matrix operand order repaired and checked by reading, not compiled, not run on hardware
//
#include <hip/hip_runtime.h>

// ---------------------------------------------------------------------------
// MI455X (gfx1250) multi-head causal self-attention, f16 WMMA / f32 accumulate
// B=4, S=2048, D=1024, H=16, HS=64.  WMMA 16x16x32 f16 throughout.
//
// Pipeline:
//  K1: fused QKV projection (double-buffered LDS W-tiles, 32x64 wave tile,
//      8 independent WMMAs per K-step).  Q pre-scaled by 1/sqrt(HS).
//      Q,K -> [B,H,S,HS] f16; V -> transposed [B,H,HS,S] f16.
//  K2: causal flash attention, one wave = 16 q rows x HS=64, online softmax,
//      batched K/V fragment loads, P re-layout via wave-private LDS.
//  K3: output projection, same GEMM structure as K1 with f16 A operand.
// ---------------------------------------------------------------------------

typedef _Float16 v16h __attribute__((ext_vector_type(16)));
typedef _Float16 v8h  __attribute__((ext_vector_type(8)));
typedef float    v8f  __attribute__((ext_vector_type(8)));

#define WMMA_F16(a, b, c) \
  __builtin_amdgcn_wmma_f32_16x16x32_f16(false, (a), false, (b), (short)0, (c), false, false)

constexpr int B_ = 4, S_ = 2048, D_ = 1024, H_ = 16, HS_ = 64;
constexpr float RSCALE = 0.125f;  // 1/sqrt(HS)

static __device__ inline v16h cat8(v8h lo, v8h hi) {
  union { v16h v; v8h h[2]; } u;
  u.h[0] = lo; u.h[1] = hi;
  return u.v;
}

// ---------------------------------------------------------------------------
// Kernel 1: fused Q/K/V projection.  Logical GEMM: [8192 x 1024] x [1024 x 3072].
// Block = 256 thr = 8 waves; block tile 256(M) x 64(N); wave tile 32 x 64.
// Double-buffered LDS W tiles: global staging of tile k+1 overlaps WMMAs on k.
// ---------------------------------------------------------------------------
__global__ __launch_bounds__(256) void qkv_proj_kernel(
    const float* __restrict__ x,
    const float* __restrict__ Wq, const float* __restrict__ bq,
    const float* __restrict__ Wk, const float* __restrict__ bk,
    const float* __restrict__ Wv, const float* __restrict__ bv,
    _Float16* __restrict__ Qh, _Float16* __restrict__ Kh,
    _Float16* __restrict__ Vt) {
  __shared__ _Float16 wt[2][64 * 40];  // [buf][n][k] transposed, padded stride 40

  const int lane = threadIdx.x & 31;
  const int wave = threadIdx.x >> 5;
  const int hl = lane >> 4;   // half-wave id (wave32)
  const int ln = lane & 15;

  const int cBase = blockIdx.x * 64;         // global col in [0, 3072)
  const int mat   = cBase >> 10;             // 0=Q 1=K 2=V
  const int ncol0 = cBase & 1023;
  const int mBase = blockIdx.y * 256 + wave * 32;

  const float* W    = (mat == 0) ? Wq : (mat == 1) ? Wk : Wv;
  const float* bias = (mat == 0) ? bq : (mat == 1) ? bk : bv;

  v8f acc[2][4] = {};  // [A-row subtile][N subtile] f32 accumulators

  const float* arow0 = x + (size_t)(mBase + ln) * D_ + hl * 16;
  const float* arow1 = arow0 + (size_t)16 * D_;

  auto stage = [&](int buf, int k0) {
#pragma unroll
    for (int i = 0; i < 8; ++i) {
      int e = threadIdx.x + i * 256;
      int n = e & 63, k = e >> 6;
      wt[buf][n * 40 + k] = (_Float16)W[(size_t)(k0 + k) * D_ + ncol0 + n];
    }
  };

  stage(0, 0);
  for (int kk = 0; kk < D_ / 32; ++kk) {
    const int k0 = kk * 32;
    const int buf = kk & 1;
    __syncthreads();                           // wt[buf] ready; wt[buf^1] free
    if (kk + 1 < D_ / 32) stage(buf ^ 1, k0 + 32);  // overlaps WMMAs below

    // Two A fragments (rows ln and 16+ln), f32 -> f16 (A layout: 16
    // contiguous halves per lane at k0+hl*16, matching B).
    v8h lo0, hi0, lo1, hi1;
#pragma unroll
    for (int i = 0; i < 8; ++i) {
      lo0[i] = (_Float16)arow0[k0 + i];
      hi0[i] = (_Float16)arow0[k0 + 8 + i];
      lo1[i] = (_Float16)arow1[k0 + i];
      hi1[i] = (_Float16)arow1[k0 + 8 + i];
    }
    const v16h a0 = cat8(lo0, hi0);
    const v16h a1 = cat8(lo1, hi1);

    // Batch all four B fragments, then a burst of 8 independent WMMAs.
    v16h b[4];
#pragma unroll
    for (int t = 0; t < 4; ++t)
      b[t] = *(const v16h*)&wt[buf][(t * 16 + ln) * 40 + hl * 16];
#pragma unroll
    for (int t = 0; t < 4; ++t) {
      acc[0][t] = WMMA_F16(a0, b[t], acc[0][t]);
      acc[1][t] = WMMA_F16(a1, b[t], acc[1][t]);
    }
  }

  // Epilogue: bias, Q-scale, f16 store into head-major layouts.
#pragma unroll
  for (int ar = 0; ar < 2; ++ar) {
#pragma unroll
    for (int t = 0; t < 4; ++t) {
      const int ncol = (cBase + t * 16 + ln) & 1023;
      const int h = ncol >> 6, d = ncol & 63;
      const float bvn = bias[ncol];
#pragma unroll
      for (int v = 0; v < 8; ++v) {
        const int m = mBase + ar * 16 + v + 8 * hl;  // C/D layout: m = v+8*half
        const int b2 = m >> 11, s = m & 2047;
        float val = acc[ar][t][v] + bvn;
        if (mat == 0) val = val * RSCALE;
        const _Float16 hv2 = (_Float16)val;
        const int bhi = b2 * H_ + h;
        if (mat == 2)
          Vt[((size_t)(bhi * HS_ + d)) * S_ + s] = hv2;   // [B,H,HS,S]
        else if (mat == 0)
          Qh[((size_t)(bhi * S_ + s)) * HS_ + d] = hv2;   // [B,H,S,HS]
        else
          Kh[((size_t)(bhi * S_ + s)) * HS_ + d] = hv2;
      }
    }
  }
}

// ---------------------------------------------------------------------------
// Kernel 2: causal flash attention.  Block = 128 thr = 4 waves, 64 q rows.
// ---------------------------------------------------------------------------
__global__ __launch_bounds__(128) void attn_kernel(
    const _Float16* __restrict__ Qh, const _Float16* __restrict__ Kh,
    const _Float16* __restrict__ Vt, _Float16* __restrict__ Ah) {
  __shared__ _Float16 pstage[4][16 * 72];  // per-wave P tile [16 q][64 keys]

  const int lane = threadIdx.x & 31;
  const int wave = threadIdx.x >> 5;
  const int hl = lane >> 4, ln = lane & 15;
  const int bh = blockIdx.y;               // b*H + h
  const int qB = blockIdx.x * 64;
  const int q0 = qB + wave * 16;

  const _Float16* Qb = Qh + (size_t)bh * S_ * HS_;
  const _Float16* Kb = Kh + (size_t)bh * S_ * HS_;
  const _Float16* Vb = Vt + (size_t)bh * HS_ * S_;

  // Persistent Q A-fragments (d = 0..31, 32..63).
  const _Float16* qrow = Qb + (size_t)(q0 + ln) * HS_ + hl * 16;
  const v16h aq0 = cat8(*(const v8h*)(qrow),      *(const v8h*)(qrow + 8));
  const v16h aq1 = cat8(*(const v8h*)(qrow + 32), *(const v8h*)(qrow + 40));

  v8f o[4] = {};              // 16 x 64 output accumulator (f32)
  float mrow[8], lrow[8];     // per-row softmax state, row = v + 8*hl
#pragma unroll
  for (int v = 0; v < 8; ++v) { mrow[v] = -1e30f; lrow[v] = 0.0f; }

  _Float16* pw = &pstage[wave][0];

  for (int kb = 0; kb <= (int)blockIdx.x; ++kb) {
    const int key0 = kb * 64;

    if (kb < (int)blockIdx.x) {  // prefetch next K/V block (global_prefetch)
      __builtin_prefetch(Kb + (size_t)(key0 + 64) * HS_ + lane * 32, 0, 1);
      __builtin_prefetch(Vb + (size_t)lane * S_ + key0 + 64, 0, 1);
    }

    // ---- scores = Q . K^T: batch all 8 K B-fragments, then 8 WMMAs ----
    v16h kb0[4], kb1[4];
#pragma unroll
    for (int kt = 0; kt < 4; ++kt) {
      const _Float16* krow = Kb + (size_t)(key0 + kt * 16 + ln) * HS_ + hl * 16;
      kb0[kt] = *(const v16h*)(krow);        // d = 0..31
      kb1[kt] = *(const v16h*)(krow + 32);   // d = 32..63
    }
    v8f sc[4];
#pragma unroll
    for (int kt = 0; kt < 4; ++kt) {
      v8f c = {};
      c = WMMA_F16(aq0, kb0[kt], c);
      sc[kt] = WMMA_F16(aq1, kb1[kt], c);
    }

    if (key0 == qB) {  // diagonal block: causal mask (key > q)
#pragma unroll
      for (int kt = 0; kt < 4; ++kt)
#pragma unroll
        for (int v = 0; v < 8; ++v)
          if (key0 + kt * 16 + ln > q0 + v + 8 * hl) sc[kt][v] = -1e30f;
    }

    // ---- online softmax: row reductions via 16-lane xor butterflies ----
    float pm[8];
#pragma unroll
    for (int v = 0; v < 8; ++v)
      pm[v] = fmaxf(fmaxf(sc[0][v], sc[1][v]), fmaxf(sc[2][v], sc[3][v]));
#pragma unroll
    for (int msk = 1; msk <= 8; msk <<= 1)
#pragma unroll
      for (int v = 0; v < 8; ++v)
        pm[v] = fmaxf(pm[v], __shfl_xor(pm[v], msk, 32));

    float psum[8];
#pragma unroll
    for (int v = 0; v < 8; ++v) {
      const float mnew = fmaxf(mrow[v], pm[v]);
      const float corr = __expf(mrow[v] - mnew);
      mrow[v] = mnew;
      lrow[v] = lrow[v] * corr;
#pragma unroll
      for (int dt = 0; dt < 4; ++dt) o[dt][v] = o[dt][v] * corr;
      float ps = 0.0f;
#pragma unroll
      for (int kt = 0; kt < 4; ++kt) {
        const float p = __expf(sc[kt][v] - mnew);
        sc[kt][v] = p;
        ps += p;
      }
      psum[v] = ps;
    }
#pragma unroll
    for (int msk = 1; msk <= 8; msk <<= 1)
#pragma unroll
      for (int v = 0; v < 8; ++v) psum[v] += __shfl_xor(psum[v], msk, 32);
#pragma unroll
    for (int v = 0; v < 8; ++v) lrow[v] += psum[v];

    // ---- re-layout P: C/D frag -> row-major LDS -> A frag (wave-private) ----
#pragma unroll
    for (int kt = 0; kt < 4; ++kt)
#pragma unroll
      for (int v = 0; v < 8; ++v)
        pw[(v + 8 * hl) * 72 + kt * 16 + ln] = (_Float16)sc[kt][v];

    const _Float16* prow = pw + ln * 72 + hl * 16;
    v16h pa0 = cat8(*(const v8h*)(prow),      *(const v8h*)(prow + 8));
    v16h pa1 = cat8(*(const v8h*)(prow + 32), *(const v8h*)(prow + 40));

    // ---- O += P . V: batch all 8 V B-fragments, then 8 WMMAs ----
    v16h vb0[4], vb1[4];
#pragma unroll
    for (int dt = 0; dt < 4; ++dt) {
      const _Float16* vrow = Vb + (size_t)(dt * 16 + ln) * S_ + key0 + hl * 16;
      vb0[dt] = *(const v16h*)(vrow);        // keys k..k+31
      vb1[dt] = *(const v16h*)(vrow + 32);   // keys k+32..k+63
    }
#pragma unroll
    for (int dt = 0; dt < 4; ++dt) {
      o[dt] = WMMA_F16(pa0, vb0[dt], o[dt]);
      o[dt] = WMMA_F16(pa1, vb1[dt], o[dt]);
    }
  }

  // ---- epilogue: normalize, store f16 context [B,S,D] ----
  const int b = bh >> 4, h = bh & 15;
#pragma unroll
  for (int v = 0; v < 8; ++v) {
    const int q = q0 + v + 8 * hl;
    const float inv = 1.0f / lrow[v];
#pragma unroll
    for (int dt = 0; dt < 4; ++dt)
      Ah[((size_t)(b * S_ + q)) * D_ + h * HS_ + dt * 16 + ln] =
          (_Float16)(o[dt][v] * inv);
  }
}

// ---------------------------------------------------------------------------
// Kernel 3: output projection  [8192 x 1024] x [1024 x 1024] + bias, f32 out.
// Same double-buffered structure as K1, A operand already f16.
// ---------------------------------------------------------------------------
__global__ __launch_bounds__(256) void out_proj_kernel(
    const _Float16* __restrict__ Ah, const float* __restrict__ Wm,
    const float* __restrict__ bm, float* __restrict__ out) {
  __shared__ _Float16 wt[2][64 * 40];

  const int lane = threadIdx.x & 31;
  const int wave = threadIdx.x >> 5;
  const int hl = lane >> 4, ln = lane & 15;
  const int nBase = blockIdx.x * 64;
  const int mBase = blockIdx.y * 256 + wave * 32;

  v8f acc[2][4] = {};
  const _Float16* arow0 = Ah + (size_t)(mBase + ln) * D_ + hl * 16;
  const _Float16* arow1 = arow0 + (size_t)16 * D_;

  auto stage = [&](int buf, int k0) {
#pragma unroll
    for (int i = 0; i < 8; ++i) {
      int e = threadIdx.x + i * 256;
      int n = e & 63, k = e >> 6;
      wt[buf][n * 40 + k] = (_Float16)Wm[(size_t)(k0 + k) * D_ + nBase + n];
    }
  };

  stage(0, 0);
  for (int kk = 0; kk < D_ / 32; ++kk) {
    const int k0 = kk * 32;
    const int buf = kk & 1;
    __syncthreads();
    if (kk + 1 < D_ / 32) stage(buf ^ 1, k0 + 32);

    const v16h a0 = cat8(*(const v8h*)(arow0 + k0), *(const v8h*)(arow0 + k0 + 8));
    const v16h a1 = cat8(*(const v8h*)(arow1 + k0), *(const v8h*)(arow1 + k0 + 8));

    v16h b[4];
#pragma unroll
    for (int t = 0; t < 4; ++t)
      b[t] = *(const v16h*)&wt[buf][(t * 16 + ln) * 40 + hl * 16];
#pragma unroll
    for (int t = 0; t < 4; ++t) {
      acc[0][t] = WMMA_F16(a0, b[t], acc[0][t]);
      acc[1][t] = WMMA_F16(a1, b[t], acc[1][t]);
    }
  }

#pragma unroll
  for (int ar = 0; ar < 2; ++ar) {
#pragma unroll
    for (int t = 0; t < 4; ++t) {
      const int n = nBase + t * 16 + ln;
      const float bias = bm[n];
#pragma unroll
      for (int v = 0; v < 8; ++v) {
        const int m = mBase + ar * 16 + v + 8 * hl;
        out[(size_t)m * D_ + n] = acc[ar][t][v] + bias;
      }
    }
  }
}

// ---------------------------------------------------------------------------
extern "C" void kernel_launch(void* const* d_in, const int* in_sizes, int n_in,
                              void* d_out, int out_size, void* d_ws,
                              size_t ws_size, hipStream_t stream) {
  (void)in_sizes; (void)n_in; (void)out_size; (void)ws_size;

  const float* x  = (const float*)d_in[0];
  const float* Wq = (const float*)d_in[1];
  const float* bq = (const float*)d_in[2];
  const float* Wk = (const float*)d_in[3];
  const float* bk = (const float*)d_in[4];
  const float* Wv = (const float*)d_in[5];
  const float* bv = (const float*)d_in[6];
  const float* Wm = (const float*)d_in[7];
  const float* bm = (const float*)d_in[8];
  float* out = (float*)d_out;

  const size_t NE = (size_t)B_ * H_ * S_ * HS_;  // 8M f16 elements each
  _Float16* Qh = (_Float16*)d_ws;   // [B,H,S,HS]
  _Float16* Kh = Qh + NE;           // [B,H,S,HS]
  _Float16* Vt = Kh + NE;           // [B,H,HS,S] (transposed)
  _Float16* Ah = Vt + NE;           // [B,S,D] context, f16

  qkv_proj_kernel<<<dim3(48, 32), 256, 0, stream>>>(x, Wq, bq, Wk, bk, Wv, bv,
                                                    Qh, Kh, Vt);
  attn_kernel<<<dim3(S_ / 64, B_ * H_), 128, 0, stream>>>(Qh, Kh, Vt, Ah);
  out_proj_kernel<<<dim3(16, 32), 256, 0, stream>>>(Ah, Wm, bm, out);
}
